// BiLSTM_CRF_30674656428071
// MI455X (gfx1250) — matrix operand order repaired and checked by reading, not compiled, not run on hardware
//
#include <hip/hip_runtime.h>
#include <hip/hip_bf16.h>

typedef __attribute__((ext_vector_type(16))) _Float16 v16h;
typedef __attribute__((ext_vector_type(8)))  _Float16 v8h;
typedef __attribute__((ext_vector_type(8)))  float    v8f;
typedef __attribute__((ext_vector_type(4)))  unsigned int u32x4;
typedef __attribute__((ext_vector_type(8)))  int      i32x8;
typedef __attribute__((ext_vector_type(4)))  int      i32x4;

union V16 { v16h v; v8h h[2]; };

// Problem constants
#define VOCAB 30000
#define EDIM  512
#define HDIM  512
#define HD    256
#define G4    1024      // 4*HD
#define TAGS  76
#define TPAD  80
#define BB    256
#define SS    128
#define MROWS 32768     // SS*BB

// ---------------------------------------------------------------------------
// Workspace layout (bytes)
// ---------------------------------------------------------------------------
#define OFF_WIH   ((size_t)0)                                   // 2*1024*512 f16
#define OFF_WHH   (OFF_WIH  + (size_t)2*G4*EDIM*2)              // 2*1024*256 f16
#define OFF_WOUT  (OFF_WHH  + (size_t)2*G4*HD*2)                // 80*512 f16
#define OFF_BIAS  (OFF_WOUT + (size_t)TPAD*HDIM*2)              // 2*1024 f32
#define OFF_BOUT  (OFF_BIAS + (size_t)2*G4*4)                   // 80 f32 (pad 512B)
#define OFF_ZIN   (OFF_BOUT + (size_t)512)                      // 2*32768*1024 f16
#define OFF_CBUF  (OFF_ZIN  + (size_t)2*MROWS*G4*2)             // 2*256*256 f32
#define OFF_HS    (OFF_CBUF + (size_t)2*BB*HD*4)                // 2*32768*256 f16
#define OFF_EM    (OFF_HS   + (size_t)2*MROWS*HD*2)             // 32768*80 f32
#define OFF_LLH   (OFF_EM   + (size_t)MROWS*TPAD*4)             // 256 f32

#define WMMA_F16(A, Bm, C) \
  __builtin_amdgcn_wmma_f32_16x16x32_f16(false, (A), false, (Bm), (short)0, (C), false, false)

// ---------------------------------------------------------------------------
// K1: weight conversion f32 -> f16, bias fuse, W_out zero-pad to 80 rows
// ---------------------------------------------------------------------------
__global__ void prep_kernel(const float* __restrict__ Wihf, const float* __restrict__ Wihb,
                            const float* __restrict__ Whhf, const float* __restrict__ Whhb,
                            const float* __restrict__ Wout, const float* __restrict__ bihf,
                            const float* __restrict__ bhhf, const float* __restrict__ bihb,
                            const float* __restrict__ bhhb, const float* __restrict__ bo,
                            _Float16* __restrict__ WihH, _Float16* __restrict__ WhhH,
                            _Float16* __restrict__ WoutH, float* __restrict__ biasC,
                            float* __restrict__ boutP, float* __restrict__ out) {
  size_t i = (size_t)blockIdx.x * 256 + threadIdx.x;
  const size_t NWIH = (size_t)G4 * EDIM, NWHH = (size_t)G4 * HD;
  if (i < 2 * NWIH) { WihH[i] = (_Float16)((i < NWIH) ? Wihf[i] : Wihb[i - NWIH]); return; }
  i -= 2 * NWIH;
  if (i < 2 * NWHH) { WhhH[i] = (_Float16)((i < NWHH) ? Whhf[i] : Whhb[i - NWHH]); return; }
  i -= 2 * NWHH;
  if (i < (size_t)TPAD * HDIM) {
    size_t t = i / HDIM;
    WoutH[i] = (t < TAGS) ? (_Float16)Wout[i] : (_Float16)0.f;
    return;
  }
  i -= (size_t)TPAD * HDIM;
  if (i < 2 * G4) {
    biasC[i] = (i < G4) ? (bihf[i] + bhhf[i]) : (bihb[i - G4] + bhhb[i - G4]);
    return;
  }
  i -= 2 * G4;
  if (i < TPAD) { boutP[i] = (i < TAGS) ? bo[i] : 0.f; return; }
  i -= TPAD;
  if (i == 0) out[0] = 0.f;
}

// ---------------------------------------------------------------------------
// TDM: stage a 64-row x 512-col f16 weight panel into LDS with row padding of
// 16B (LDS row stride 1040B = 520 halves -> conflict-free ds_load_b128).
// ---------------------------------------------------------------------------
#define BSTR 520  // halves per padded LDS row

__device__ __forceinline__ void tdm_stage_panel(const _Float16* gsrc, int rows) {
  unsigned long long ga = (unsigned long long)(const void*)gsrc;
  // group0: count=1, lds_addr=0, global_addr, type=2
  u32x4 g0 = { 1u, 0u, (unsigned)(ga & 0xffffffffu),
               (unsigned)((ga >> 32) & 0x1ffffffu) | 0x80000000u };
  // group1: data_size=2B, pad_enable, pad_interval=256 dw, pad_amount=4 dw,
  // tensor_dim0=512, tensor_dim1=rows, tile_dim0=512, tile_dim1=rows,
  // tensor_dim0_stride=512
  i32x8 g1 = { (int)0x07D10000,
               (int)(512u << 16),
               (int)((unsigned)rows << 16),
               (int)(512u << 16),
               rows,
               512, 0, 0 };
  i32x4 gz = { 0, 0, 0, 0 };
#if __has_include(<hip/amd_detail/amd_gfx1250_TDM.h>)
  i32x8 gz8 = { 0, 0, 0, 0, 0, 0, 0, 0 };
  __builtin_amdgcn_tensor_load_to_lds(g0, g1, gz, gz, gz8, 0);
#else
  __builtin_amdgcn_tensor_load_to_lds(g0, g1, gz, gz, 0);
#endif
  __builtin_amdgcn_s_wait_tensorcnt(0);
}

// ---------------------------------------------------------------------------
// K2: Z_in[d][m][n] = sum_k embed[ids[m]][k] * Wih[d][n][k]   (f32 acc -> f16)
// Embedding gather fused into the WMMA A-operand fetch; B panel staged in LDS
// by the Tensor Data Mover (one tensor_load_to_lds per block).
// grid (MROWS/128, G4/64, 2), block 256 (8 waves). Wave: 16 rows x 64 cols.
// ---------------------------------------------------------------------------
__global__ void zin_gemm(const int* __restrict__ ids, const float* __restrict__ embed,
                         const _Float16* __restrict__ WihH, _Float16* __restrict__ Zin) {
  extern __shared__ _Float16 ldsW[];   // 64 rows * 520 halves
  const int tid  = threadIdx.x;
  const int lane = tid & 31, wave = tid >> 5;
  const int r = lane & 15, hq = lane >> 4;
  const int d = blockIdx.z;
  const int mTile = blockIdx.x * 128 + wave * 16;
  const int nBase = blockIdx.y * 64;

  const _Float16* W = WihH + (size_t)d * G4 * EDIM;
  _Float16* Z = Zin + (size_t)d * MROWS * G4;

  if (wave == 0) tdm_stage_panel(W + (size_t)nBase * EDIM, 64);
  __syncthreads();

  // per-lane A row: m = mTile + r ; (S,B) row ordering: m = s*B + b
  const int m = mTile + r;
  const int s = m >> 8, b = m & 255;
  const float* xrow = embed + (size_t)ids[b * SS + s] * EDIM;

  v8f acc[4] = {};
  for (int kt = 0; kt < 16; ++kt) {
    const int k0 = kt * 32 + hq * 8;
    const float4 f0 = *(const float4*)(xrow + k0);
    const float4 f1 = *(const float4*)(xrow + k0 + 4);
    const float4 f2 = *(const float4*)(xrow + k0 + 16);
    const float4 f3 = *(const float4*)(xrow + k0 + 20);
    V16 a;
    v8h lo = { (_Float16)f0.x, (_Float16)f0.y, (_Float16)f0.z, (_Float16)f0.w,
               (_Float16)f1.x, (_Float16)f1.y, (_Float16)f1.z, (_Float16)f1.w };
    v8h hv = { (_Float16)f2.x, (_Float16)f2.y, (_Float16)f2.z, (_Float16)f2.w,
               (_Float16)f3.x, (_Float16)f3.y, (_Float16)f3.z, (_Float16)f3.w };
    a.h[0] = lo; a.h[1] = hv;
#pragma unroll
    for (int nt = 0; nt < 4; ++nt) {
      const _Float16* bp = ldsW + (size_t)(nt * 16 + r) * BSTR + kt * 32 + hq * 8;
      V16 bm;
      bm.h[0] = *(const v8h*)bp;
      bm.h[1] = *(const v8h*)(bp + 16);
      acc[nt] = WMMA_F16(a.v, bm.v, acc[nt]);
    }
  }
#pragma unroll
  for (int nt = 0; nt < 4; ++nt)
#pragma unroll
    for (int i = 0; i < 8; ++i)
      Z[(size_t)(mTile + i + 8 * hq) * G4 + nBase + nt * 16 + r] = (_Float16)acc[nt][i];
}

// ---------------------------------------------------------------------------
// K3: sequential LSTM, one workgroup per direction (1024 thr = 32 waves).
// h double-buffered in LDS (f16, padded rows); each wave owns fixed
// (batch-tile, j-tile) patches and computes all 4 gate tiles so the gate
// nonlinearity consumes WMMA accumulators straight from VGPRs.
// ---------------------------------------------------------------------------
#define HSTR   264            // 256 + 8 halves padding (528B rows)
#define HBUFSZ (BB * HSTR)    // halves per h buffer

__global__ void lstm_seq(const _Float16* __restrict__ WhhA, const float* __restrict__ biasA,
                         const _Float16* __restrict__ ZinA, float* __restrict__ cbufA,
                         _Float16* __restrict__ hsA, const float* __restrict__ h0,
                         const float* __restrict__ c0) {
  extern __shared__ _Float16 hb[];   // 2 * HBUFSZ halves = 264 KB
  const int d = blockIdx.x;
  const int tid  = threadIdx.x;
  const int lane = tid & 31, wave = tid >> 5;
  const int r = lane & 15, hq = lane >> 4;

  const _Float16* Whh = WhhA + (size_t)d * G4 * HD;
  const float*    bias = biasA + (size_t)d * G4;
  const _Float16* Zin  = ZinA + (size_t)d * MROWS * G4;
  float*    cb = cbufA + (size_t)d * BB * HD;
  _Float16* hs = hsA   + (size_t)d * MROWS * HD;
  const float* h0d = h0 + (size_t)d * BB * HD;
  const float* c0d = c0 + (size_t)d * BB * HD;

  // preload h0 into LDS buffer 0 as f16
  for (int e = tid; e < BB * HD; e += 1024) {
    const int bb = e >> 8, j = e & 255;
    hb[bb * HSTR + j] = (_Float16)h0d[e];
  }
  __syncthreads();

  const int mt    = wave & 15;   // batch tile (rows mt*16..+15)
  const int jbase = wave >> 4;   // j-tile parity

  for (int it = 0; it < SS; ++it) {
    const int sf = d ? (SS - 1 - it) : it;
    const size_t mrow = (size_t)sf * BB;
    const _Float16* hcur = hb + (it & 1) * HBUFSZ;
    _Float16*       hnxt = hb + ((it & 1) ^ 1) * HBUFSZ;

    // prefetch next step's 512KB Zin slice into L2 behind the WMMA chain
    if (it + 1 < SS) {
      const int sfn = d ? (SS - 2 - it) : (it + 1);
      const char* pz = (const char*)(Zin + (size_t)sfn * BB * G4);
      __builtin_prefetch(pz + (size_t)tid * 512, 0, 1);
      __builtin_prefetch(pz + (size_t)tid * 512 + 256, 0, 1);
    }

    const _Float16* arow = hcur + (mt * 16 + r) * HSTR;

    for (int g = 0; g < 8; ++g) {
      const int jt = jbase + 2 * g;          // j-tile 0..15
      v8f acc0 = {}, acc1 = {}, acc2 = {}, acc3 = {};
#pragma unroll
      for (int kt = 0; kt < 8; ++kt) {
        V16 a;
        const int k0 = kt * 32 + hq * 16;
        a.h[0] = *(const v8h*)(arow + k0);
        a.h[1] = *(const v8h*)(arow + k0 + 8);
        const size_t wro = (size_t)kt * 32 + hq * 16;
        const v16h b0 = *(const v16h*)(Whh + (size_t)((jt)      * 16 + r) * HD + wro);
        const v16h b1 = *(const v16h*)(Whh + (size_t)((16 + jt) * 16 + r) * HD + wro);
        const v16h b2 = *(const v16h*)(Whh + (size_t)((32 + jt) * 16 + r) * HD + wro);
        const v16h b3 = *(const v16h*)(Whh + (size_t)((48 + jt) * 16 + r) * HD + wro);
        acc0 = WMMA_F16(a.v, b0, acc0);
        acc1 = WMMA_F16(a.v, b1, acc1);
        acc2 = WMMA_F16(a.v, b2, acc2);
        acc3 = WMMA_F16(a.v, b3, acc3);
      }
      // gate phase straight out of the accumulators
      const int j = jt * 16 + r;
#pragma unroll
      for (int i = 0; i < 8; ++i) {
        const int brow = mt * 16 + i + 8 * hq;
        const size_t zi = (mrow + brow) * (size_t)G4;
        const float z_i = acc0[i] + (float)Zin[zi + j]            + bias[j];
        const float z_f = acc1[i] + (float)Zin[zi + HD + j]       + bias[HD + j];
        const float z_g = acc2[i] + (float)Zin[zi + 2 * HD + j]   + bias[2 * HD + j];
        const float z_o = acc3[i] + (float)Zin[zi + 3 * HD + j]   + bias[3 * HD + j];
        const int ce = brow * HD + j;
        const float cprev = (it == 0) ? c0d[ce] : cb[ce];
        const float ig = 1.f / (1.f + __expf(-z_i));
        const float fg = 1.f / (1.f + __expf(-z_f));
        const float og = 1.f / (1.f + __expf(-z_o));
        const float gg = tanhf(z_g);
        const float cn = fg * cprev + ig * gg;
        const float hn = og * tanhf(cn);
        cb[ce] = cn;
        hnxt[brow * HSTR + j] = (_Float16)hn;
        hs[(mrow + brow) * HD + j] = (_Float16)hn;
      }
    }
    __syncthreads();
  }
}

// ---------------------------------------------------------------------------
// K4: emissions: em[m][t] = [hf|hb][m] . W_out[t] + b_out[t], N padded to 80.
// grid 256, block 256 (8 waves); wave: 16 rows x 5 n-tiles, A reused over N.
// ---------------------------------------------------------------------------
__global__ void em_gemm(const _Float16* __restrict__ hsA, const _Float16* __restrict__ WoutH,
                        const float* __restrict__ boutP, float* __restrict__ em) {
  const int tid  = threadIdx.x;
  const int lane = tid & 31, wave = tid >> 5;
  const int r = lane & 15, hq = lane >> 4;
  const int mTile = blockIdx.x * 128 + wave * 16;

  const _Float16* arowf = hsA + (size_t)(mTile + r) * HD;
  const _Float16* arowb = hsA + (size_t)MROWS * HD + (size_t)(mTile + r) * HD;

  v8f acc[5] = {};
  for (int kt = 0; kt < 16; ++kt) {
    const _Float16* src = (kt < 8) ? (arowf + kt * 32) : (arowb + (kt - 8) * 32);
    V16 a;
    a.h[0] = *(const v8h*)(src + hq * 16);
    a.h[1] = *(const v8h*)(src + hq * 16 + 8);
#pragma unroll
    for (int nt = 0; nt < 5; ++nt) {
      const v16h bm = *(const v16h*)(WoutH + (size_t)(nt * 16 + r) * HDIM + kt * 32 + hq * 16);
      acc[nt] = WMMA_F16(a.v, bm, acc[nt]);
    }
  }
#pragma unroll
  for (int nt = 0; nt < 5; ++nt)
#pragma unroll
    for (int i = 0; i < 8; ++i)
      em[(size_t)(mTile + i + 8 * hq) * TPAD + nt * 16 + r] = acc[nt][i] + boutP[nt * 16 + r];
}

// ---------------------------------------------------------------------------
// K5: CRF NLL per batch element (block = 128 threads, 76 active tags)
// ---------------------------------------------------------------------------
__global__ void crf_nll(const float* __restrict__ em, const int* __restrict__ tag_ids,
                        const int* __restrict__ lengths, const float* __restrict__ stt,
                        const float* __restrict__ ett, const float* __restrict__ trans,
                        float* __restrict__ llh) {
  const int b = blockIdx.x;
  const int t = threadIdx.x;
  __shared__ float alpha[TPAD];
  __shared__ float gold[1];
  const int len = lengths[b];

  if (t < TAGS) alpha[t] = stt[t] + em[(size_t)b * TPAD + t];
  if (t == 0) {
    int prev = tag_ids[b * SS];
    float score = stt[prev] + em[(size_t)b * TPAD + prev];
    for (int s = 1; s < SS; ++s) {
      if (s < len) {
        const int tg = tag_ids[b * SS + s];
        score += trans[prev * TAGS + tg] + em[((size_t)s * BB + b) * TPAD + tg];
        prev = tg;
      }
    }
    gold[0] = score + ett[prev];
  }
  __syncthreads();

  for (int s = 1; s < SS; ++s) {
    float nv = 0.f;
    if (t < TAGS && s < len) {
      float mx = -3.4e38f;
      for (int p = 0; p < TAGS; ++p)
        mx = fmaxf(mx, alpha[p] + trans[p * TAGS + t]);
      float sum = 0.f;
      for (int p = 0; p < TAGS; ++p)
        sum += __expf(alpha[p] + trans[p * TAGS + t] - mx);
      nv = mx + __logf(sum) + em[((size_t)s * BB + b) * TPAD + t];
    }
    __syncthreads();
    if (t < TAGS && s < len) alpha[t] = nv;
    __syncthreads();
  }

  if (t == 0) {
    float mx = -3.4e38f;
    for (int p = 0; p < TAGS; ++p) mx = fmaxf(mx, alpha[p] + ett[p]);
    float sum = 0.f;
    for (int p = 0; p < TAGS; ++p) sum += __expf(alpha[p] + ett[p] - mx);
    llh[b] = gold[0] - (mx + __logf(sum));
  }
}

// ---------------------------------------------------------------------------
// K6: deterministic mean reduction: out = -mean(llh)
// ---------------------------------------------------------------------------
__global__ void reduce_mean(const float* __restrict__ llh, float* __restrict__ out) {
  __shared__ float sm[256];
  const int t = threadIdx.x;
  sm[t] = llh[t];
  __syncthreads();
  for (int off = 128; off > 0; off >>= 1) {
    if (t < off) sm[t] += sm[t + off];
    __syncthreads();
  }
  if (t == 0) out[0] = -sm[0] / 256.0f;
}

// ---------------------------------------------------------------------------
extern "C" void kernel_launch(void* const* d_in, const int* in_sizes, int n_in,
                              void* d_out, int out_size, void* d_ws, size_t ws_size,
                              hipStream_t stream) {
  const int*   ids   = (const int*)d_in[0];
  const int*   tags  = (const int*)d_in[1];
  const int*   lens  = (const int*)d_in[2];
  const float* embed = (const float*)d_in[3];
  const float* Wihf  = (const float*)d_in[4];
  const float* Whhf  = (const float*)d_in[5];
  const float* bihf  = (const float*)d_in[6];
  const float* bhhf  = (const float*)d_in[7];
  const float* Wihb  = (const float*)d_in[8];
  const float* Whhb  = (const float*)d_in[9];
  const float* bihb  = (const float*)d_in[10];
  const float* bhhb  = (const float*)d_in[11];
  const float* Wout  = (const float*)d_in[12];
  const float* bo    = (const float*)d_in[13];
  const float* stt   = (const float*)d_in[14];
  const float* ett   = (const float*)d_in[15];
  const float* trans = (const float*)d_in[16];
  const float* h0    = (const float*)d_in[17];
  const float* c0    = (const float*)d_in[18];

  char* ws = (char*)d_ws;
  _Float16* WihH  = (_Float16*)(ws + OFF_WIH);
  _Float16* WhhH  = (_Float16*)(ws + OFF_WHH);
  _Float16* WoutH = (_Float16*)(ws + OFF_WOUT);
  float*    biasC = (float*)(ws + OFF_BIAS);
  float*    boutP = (float*)(ws + OFF_BOUT);
  _Float16* Zin   = (_Float16*)(ws + OFF_ZIN);
  float*    cbuf  = (float*)(ws + OFF_CBUF);
  _Float16* hsB   = (_Float16*)(ws + OFF_HS);
  float*    emB   = (float*)(ws + OFF_EM);
  float*    llh   = (float*)(ws + OFF_LLH);
  float*    outF  = (float*)d_out;

  // K1: weight conversion (covers 1,615,953 work items)
  prep_kernel<<<6313, 256, 0, stream>>>(Wihf, Wihb, Whhf, Whhb, Wout, bihf, bhhf,
                                        bihb, bhhb, bo, WihH, WhhH, WoutH, biasC,
                                        boutP, outF);
  // K2: big time-parallel input-projection GEMM (TDM-staged B panel in LDS)
  zin_gemm<<<dim3(MROWS / 128, G4 / 64, 2), 256, 64 * BSTR * sizeof(_Float16), stream>>>(
      ids, embed, WihH, Zin);
  // K3: sequential recurrence, one persistent WG per direction, 264KB LDS
  lstm_seq<<<2, 1024, 2 * HBUFSZ * sizeof(_Float16), stream>>>(WhhH, biasC, Zin,
                                                               cbuf, hsB, h0, c0);
  // K4: emission GEMM
  em_gemm<<<MROWS / 128, 256, 0, stream>>>(hsB, WoutH, boutP, emB);
  // K5: CRF forward + gold score per batch element
  crf_nll<<<BB, 128, 0, stream>>>(emB, tags, lens, stt, ett, trans, llh);
  // K6: deterministic mean
  reduce_mean<<<1, 256, 0, stream>>>(llh, outF);
}
